// GlimpseNetwork_3959959847126
// MI455X (gfx1250) — matrix operand order repaired and checked by reading, not compiled, not run on hardware
//
#include <hip/hip_runtime.h>
#include <hip/hip_bf16.h>

// GlimpseNetwork forward for MI455X (gfx1250, wave32, WMMA).
// v2: weights pre-converted f32->bf16 into WMMA B-fragment order (one-shot
// swizzle kernel), so the GEMM inner loop is barrier-free and LDS-free:
// 2x global_load_b128 (A) + 2x global_load_b128 (B) + 1x v_wmma per K=32.
// Everything (W1 bf16 = 18.9 MB, W3|W4 bf16 = 8.4 MB, activations) is
// L2-resident (192 MB), so cross-block reuse needs no LDS staging.

typedef __attribute__((ext_vector_type(16))) __bf16 v16bf;
typedef __attribute__((ext_vector_type(8)))  float  v8f;

union Frag16 { unsigned int u[8]; v16bf v; };

__device__ __forceinline__ unsigned short f2bf(float f) {
    // round-to-nearest-even f32 -> bf16
    unsigned int u = __builtin_bit_cast(unsigned int, f);
    u += 0x7FFFu + ((u >> 16) & 1u);
    return (unsigned short)(u >> 16);
}

// ---------------------------------------------------------------------------
// Foveation: one thread per glimpse element (b, s*3072 + c*1024 + i*32 + j).
// Patch size = 32<<s, pool kernel = 1<<s, pad = size/2+1; zero outside image.
// Output glimpse (256, 9216) row-major bf16 (direct WMMA A-matrix feed).
// ---------------------------------------------------------------------------
__global__ void foveate_kernel(const float* __restrict__ x,
                               const float* __restrict__ l_prev,
                               unsigned short* __restrict__ glimpse) {
    int tid = blockIdx.x * 256 + threadIdx.x;
    if (tid >= 256 * 9216) return;
    int b   = tid / 9216;
    int r   = tid - b * 9216;
    int s   = r / 3072;
    int rem = r - s * 3072;
    int c   = rem / 1024;
    rem    -= c * 1024;
    int i   = rem >> 5;
    int j   = rem & 31;

    float l0 = l_prev[b * 2 + 0];      // col coordinate
    float l1 = l_prev[b * 2 + 1];      // row coordinate
    int st0 = (int)(0.5f * ((l0 + 1.0f) * 224.0f));
    int st1 = (int)(0.5f * ((l1 + 1.0f) * 224.0f));

    int k    = 1 << s;
    int size = 32 << s;
    int pad  = (size >> 1) + 1;
    int row0 = st1 - pad + i * k;
    int col0 = st0 - pad + j * k;

    const float* xp = x + (size_t)(b * 3 + c) * 224 * 224;
    float sum = 0.0f;
    for (int u = 0; u < k; ++u) {
        int row = row0 + u;
        if (row < 0 || row >= 224) continue;
        const float* xr = xp + row * 224;
        for (int v = 0; v < k; ++v) {
            int col = col0 + v;
            if (col >= 0 && col < 224) sum += xr[col];
        }
    }
    glimpse[(size_t)b * 9216 + r] = f2bf(sum / (float)(k * k));
}

// ---------------------------------------------------------------------------
// l_out = relu(l_prev @ W2 + b2) -> bf16 into columns [1024,2048) of A2.
// ---------------------------------------------------------------------------
__global__ void locnet_kernel(const float* __restrict__ l_prev,
                              const float* __restrict__ W2,
                              const float* __restrict__ b2,
                              unsigned short* __restrict__ A2) {
    int tid = blockIdx.x * 256 + threadIdx.x;
    if (tid >= 256 * 1024) return;
    int b = tid >> 10;
    int n = tid & 1023;
    float v = l_prev[b * 2 + 0] * W2[n] + l_prev[b * 2 + 1] * W2[1024 + n] + b2[n];
    v = fmaxf(v, 0.0f);
    A2[(size_t)b * 2048 + 1024 + n] = f2bf(v);
}

// ---------------------------------------------------------------------------
// Weight swizzle: f32 (K,N) row-major -> bf16 in WMMA B-fragment order.
// For K-block kb (32 rows) and column n, store 16 dwords:
//   Bswz[((kb*N + n)*16) + rp] = pack(bf16 W[kb*32+2rp][n], bf16 W[kb*32+2rp+1][n])
// rp = jj + 8*half gives contiguous k order, same as A (VGPR jj holds K={2jj,2jj+1}
// for lanes 0-15, K=16+{2jj,2jj+1} for lanes 16-31). Rows k >= splitK come
// from Wb (fuses the W3;W4 stack for GEMM2).
// One thread per (kb, n): reads are lane-coalesced per row, writes are 64B
// contiguous per thread.
// ---------------------------------------------------------------------------
__global__ void swizzle_weights_kernel(const float* __restrict__ Wa,
                                       const float* __restrict__ Wb,
                                       int splitK, int N, int Kblocks,
                                       unsigned int* __restrict__ Bswz) {
    int tid = blockIdx.x * 256 + threadIdx.x;
    if (tid >= Kblocks * N) return;
    int kb = tid / N;
    int n  = tid - kb * N;
    int k0 = kb * 32;
    const float* W = (k0 < splitK) ? Wa : Wb;
    int kr = (k0 < splitK) ? k0 : (k0 - splitK);
    unsigned int* dst = Bswz + ((size_t)kb * N + n) * 16;
#pragma unroll
    for (int rp = 0; rp < 16; ++rp) {
        unsigned int lo = f2bf(W[(size_t)(kr + 2 * rp) * N + n]);
        unsigned int hi = f2bf(W[(size_t)(kr + 2 * rp + 1) * N + n]);
        dst[rp] = lo | (hi << 16);
    }
}

// ---------------------------------------------------------------------------
// Barrier-free WMMA GEMM: out = relu(A_bf16 @ Bswz + bias0 [+ bias1]).
// Block = 256 threads = 8 wave32s; each wave owns one 16(M) x 16(N) tile;
// block tile = 16 x 128. Per K-step of 32: A frag = 2x b128 (row-major),
// B frag = 2x b128 (pre-swizzled), one v_wmma_f32_16x16x32_bf16.
// ---------------------------------------------------------------------------
__global__ void wmma_gemm_kernel(const unsigned short* __restrict__ A, int lda,
                                 const unsigned int* __restrict__ Bswz, int N,
                                 const float* __restrict__ bias0,
                                 const float* __restrict__ bias1,
                                 int Kblocks,
                                 void* __restrict__ outp, int ldo, int outBf16) {
    const int tid  = threadIdx.x;
    const int wave = tid >> 5;
    const int lane = tid & 31;
    const int half = lane >> 4;
    const int mrow = lane & 15;
    const int m0 = blockIdx.y * 16;
    const int n  = blockIdx.x * 128 + wave * 16 + (lane & 15);

    const unsigned int* arow =
        (const unsigned int*)(A + (size_t)(m0 + mrow) * lda);
    const unsigned int* bp = Bswz + (size_t)n * 16 + half * 8;
    const size_t bstep = (size_t)N * 16;          // dwords per K-block

    v8f acc = {};
    for (int kb = 0; kb < Kblocks; ++kb) {
        Frag16 af, bf;
        const unsigned int* ak = arow + (size_t)kb * 16 + half * 8;
#pragma unroll
        for (int jj = 0; jj < 8; ++jj)            // 8 contiguous dwords + half*8
            af.u[jj] = ak[jj];
#pragma unroll
        for (int jj = 0; jj < 8; ++jj)            // 8 contiguous dwords
            bf.u[jj] = bp[jj];
        // stream-prefetch B four K-blocks ahead (speculative: OOB dropped)
        __builtin_prefetch(bp + 4 * bstep, 0, 1);
        acc = __builtin_amdgcn_wmma_f32_16x16x32_bf16(
            /*neg_a=*/false, af.v, /*neg_b=*/false, bf.v,
            /*c_mod=*/(short)0, acc, /*reuse_a=*/false, /*reuse_b=*/false);
        bp += bstep;
    }

    // D layout: VGPR r -> M = r + 8*half, N = lane&15 within the wave tile.
    float bsum = bias0[n] + (bias1 ? bias1[n] : 0.0f);
#pragma unroll
    for (int rr = 0; rr < 8; ++rr) {
        int m = m0 + rr + half * 8;
        float v = acc[rr] + bsum;
        v = fmaxf(v, 0.0f);
        if (outBf16)
            ((unsigned short*)outp)[(size_t)m * ldo + n] = f2bf(v);
        else
            ((float*)outp)[(size_t)m * ldo + n] = v;
    }
}

// ---------------------------------------------------------------------------
extern "C" void kernel_launch(void* const* d_in, const int* in_sizes, int n_in,
                              void* d_out, int out_size, void* d_ws, size_t ws_size,
                              hipStream_t stream) {
    const float* x      = (const float*)d_in[0];
    const float* l_prev = (const float*)d_in[1];
    const float* W1 = (const float*)d_in[2];
    const float* b1 = (const float*)d_in[3];
    const float* W2 = (const float*)d_in[4];
    const float* b2 = (const float*)d_in[5];
    const float* W3 = (const float*)d_in[6];
    const float* b3 = (const float*)d_in[7];
    const float* W4 = (const float*)d_in[8];
    const float* b4 = (const float*)d_in[9];
    float* out = (float*)d_out;

    // Workspace layout (bytes):
    //   glimpse bf16 256x9216   : 4,718,592
    //   A2      bf16 256x2048   : 1,048,576
    //   Bswz1   bf16 9216x1024  : 18,874,368  (288 K-blocks)
    //   Bswz2   bf16 2048x2048  :  8,388,608  (64 K-blocks)   total ~31.5 MB
    unsigned short* glimpse = (unsigned short*)d_ws;
    unsigned short* A2      = glimpse + (size_t)256 * 9216;
    unsigned int*   Bswz1   = (unsigned int*)(A2 + (size_t)256 * 2048);
    unsigned int*   Bswz2   = Bswz1 + (size_t)288 * 1024 * 16;

    // Weight swizzles (independent of activations)
    swizzle_weights_kernel<<<(288 * 1024) / 256, 256, 0, stream>>>(
        W1, W1, /*splitK=*/9216, /*N=*/1024, /*Kblocks=*/288, Bswz1);
    swizzle_weights_kernel<<<(64 * 2048) / 256, 256, 0, stream>>>(
        W3, W4, /*splitK=*/1024, /*N=*/2048, /*Kblocks=*/64, Bswz2);

    // 1) foveate -> glimpse bf16
    foveate_kernel<<<9216, 256, 0, stream>>>(x, l_prev, glimpse);

    // 2) l_out -> A2[:, 1024:2048]
    locnet_kernel<<<1024, 256, 0, stream>>>(l_prev, W2, b2, A2);

    // 3) g_out = relu(glimpse @ W1 + b1) -> A2[:, 0:1024] bf16
    wmma_gemm_kernel<<<dim3(1024 / 128, 256 / 16), 256, 0, stream>>>(
        glimpse, 9216, Bswz1, 1024, b1, nullptr, 288, A2, 2048, 1);

    // 4) out = relu([g_out|l_out] @ [W3;W4] + b3 + b4), f32
    wmma_gemm_kernel<<<dim3(2048 / 128, 256 / 16), 256, 0, stream>>>(
        A2, 2048, Bswz2, 2048, b3, b4, 64, out, 2048, 0);
}
